// PhysicsAttentionBase_88021059764296
// MI455X (gfx1250) — matrix operand order repaired and checked by reading, not compiled, not run on hardware
//
#include <hip/hip_runtime.h>

// ---------- types ----------
typedef __attribute__((ext_vector_type(16))) __bf16 bf16x16;
typedef __attribute__((ext_vector_type(8)))  float  floatx8;
typedef __attribute__((ext_vector_type(4)))  unsigned int v4u;
typedef __attribute__((ext_vector_type(4)))  int  v4i;
typedef __attribute__((ext_vector_type(8)))  int  v8i;

struct alignas(16) f4_t { float x, y, z, w; };
struct alignas(8)  u2_t { unsigned x, y; };

union FragAB { bf16x16 v; f4_t f4[2]; };

__device__ __forceinline__ unsigned short f2bf(float f) {
  union { float f; unsigned u; } v; v.f = f;
  unsigned r = v.u + 0x7FFFu + ((v.u >> 16) & 1u);
  return (unsigned short)(r >> 16);
}

#define WMMA_BF16(A, B, C) \
  __builtin_amdgcn_wmma_f32_16x16x32_bf16(false, (A), false, (B), (short)0, (C), false, false)

// Byte offset of an LDS object: flat LDS aperture keeps the offset in addr[31:0].
#define LDS_OFF(p) ((unsigned)(size_t)(p))

// ---------------------------------------------------------------------------
// TDM: async 2D strided tile (bf16 elements) -> LDS, packed rows.
// D# per CDNA5 ISA: group0 {count, lds_addr, global_addr, type=2},
// group1 {data_size=1(2B), tensor/tile dims, dim0 stride}.  Issued per-wave.
// ---------------------------------------------------------------------------
__device__ __forceinline__ void tdm_load_tile_bf16(unsigned lds_off, const void* gptr,
                                                   unsigned tile_w, unsigned tile_h,
                                                   unsigned stride_elems)
{
  unsigned long long ga = (unsigned long long)(size_t)gptr;
  v4u g0;
  g0[0] = 1u;                                         // count=1, user mode
  g0[1] = lds_off;                                    // LDS byte address
  g0[2] = (unsigned)(ga & 0xffffffffu);               // global addr [95:64]
  g0[3] = (unsigned)((ga >> 32) & 0x01ffffffu) | 0x80000000u;  // addr hi | type=2
  v8i g1;
  g1[0] = (int)(1u << 16);                            // data_size=1 (2 bytes)
  g1[1] = (int)((stride_elems & 0xffffu) << 16);      // tensor_dim0 lo16 @ [63:48]
  g1[2] = (int)((stride_elems >> 16) | (tile_h << 16)); // dim0 hi16 | tensor_dim1 lo16
  g1[3] = (int)((tile_h >> 16) | (tile_w << 16));     // tensor_dim1 hi16 | tile_dim0
  g1[4] = (int)(tile_h & 0xffffu);                    // tile_dim1 | tile_dim2(0)<<16
  g1[5] = (int)stride_elems;                          // tensor_dim0_stride lo32
  g1[6] = 0;                                          // stride0 hi16 | stride1 lo16
  g1[7] = 0;
  v4i z4 = {0, 0, 0, 0};
#if __clang_major__ >= 23
  v8i z8 = {0, 0, 0, 0, 0, 0, 0, 0};
  __builtin_amdgcn_tensor_load_to_lds(g0, g1, z4, z4, z8, 0);
#else
  __builtin_amdgcn_tensor_load_to_lds(g0, g1, z4, z4, 0);
#endif
}

// Problem constants: B=2, N=32768, C=512, H=8, D=64, S=64, rows=B*N=65536.

// ---------------------------------------------------------------------------
// K0: prep. Wps[c,(h,s)] = sum_d Wx[c,(h,d)]*Wslice[d,s] (bf16); Wfx -> bf16;
//     bps = bx_h@Wslice + bslice; snorm=0.01 (epsilon folded); stok=0.
// ---------------------------------------------------------------------------
__global__ __launch_bounds__(512)
void k0_prep(const float* __restrict__ Wx, const float* __restrict__ Wfx,
             const float* __restrict__ Wslice, const float* __restrict__ bx,
             const float* __restrict__ bslice,
             unsigned short* __restrict__ Wfx_bf, unsigned short* __restrict__ Wps_bf,
             float* __restrict__ bps, float* __restrict__ snorm,
             float* __restrict__ stok)
{
  const int c = blockIdx.x, j = threadIdx.x;
  const int h = j >> 6, s = j & 63;
  float acc = 0.f;
#pragma unroll 8
  for (int d = 0; d < 64; ++d)
    acc += Wx[(size_t)c * 512 + h * 64 + d] * Wslice[d * 64 + s];
  Wps_bf[(size_t)c * 512 + j] = f2bf(acc);
  Wfx_bf[(size_t)c * 512 + j] = f2bf(Wfx[(size_t)c * 512 + j]);
  if (c == 0) {
    float b = bslice[s];
#pragma unroll 8
    for (int d = 0; d < 64; ++d) b += bx[h * 64 + d] * Wslice[d * 64 + s];
    bps[j] = b;
  }
  if (c < 128)  stok[(size_t)c * 512 + j] = 0.f;
  if (c == 128) snorm[j] = 0.01f;
  if (c == 129) snorm[512 + j] = 0.01f;
}

// ---------------------------------------------------------------------------
// K1: fused GEMM [65536x512] x [512x1024], BM=128 BN=128 BK=64, 8 waves (4x2).
//   cols 0..511   -> fx bf16 (+bfx)
//   cols 512..1023-> temperature softmax over 64-col head groups IN-EPILOGUE
//                    -> sw bf16 + atomic slice-mass into snorm.
// B tile staged by TDM (wave 0), A tile f32->bf16 by VALU in parallel.
// ---------------------------------------------------------------------------
__global__ __launch_bounds__(256)
void k1_gemm_fused(const float* __restrict__ x,
                   const unsigned short* __restrict__ Wfx_bf,
                   const unsigned short* __restrict__ Wps_bf,
                   const float* __restrict__ bfx, const float* __restrict__ bps,
                   const float* __restrict__ temperature,
                   unsigned short* __restrict__ fx_out,
                   unsigned short* __restrict__ sw_out,
                   float* __restrict__ snorm)
{
  __shared__ alignas(16) unsigned short As[128 * 64];  // [m][k]
  __shared__ alignas(16) unsigned short Bs[64 * 128];  // [k][n]
  const int tid = threadIdx.x, lane = tid & 31, wave = tid >> 5;
  const int wrow = wave & 3, wcol = wave >> 2;
  const int bm = blockIdx.x * 128;
  const int bn = blockIdx.y * 128;                     // 0..1023
  const int b  = bm >> 15;
  const unsigned short* Bsrc = (bn < 512) ? (Wfx_bf + bn) : (Wps_bf + (bn - 512));

  floatx8 acc[2][4];
#pragma unroll
  for (int t = 0; t < 2; ++t)
#pragma unroll
    for (int j = 0; j < 4; ++j)
#pragma unroll
      for (int r = 0; r < 8; ++r) acc[t][j][r] = 0.f;

  for (int k0 = 0; k0 < 512; k0 += 64) {
    if (wave == 0)
      tdm_load_tile_bf16(LDS_OFF(Bs), Bsrc + (size_t)k0 * 512, 128, 64, 512);
    // A tile: 128x64 f32 -> bf16 LDS (overlaps the TDM)
#pragma unroll
    for (int i = 0; i < 8; ++i) {
      int idx = tid + i * 256;
      int r = idx >> 4, c4 = (idx & 15) << 2;
      f4_t v = *(const f4_t*)(x + (size_t)(bm + r) * 512 + k0 + c4);
      u2_t p;
      p.x = (unsigned)f2bf(v.x) | ((unsigned)f2bf(v.y) << 16);
      p.y = (unsigned)f2bf(v.z) | ((unsigned)f2bf(v.w) << 16);
      *(u2_t*)(&As[r * 64 + c4]) = p;
    }
    if (wave == 0) __builtin_amdgcn_s_wait_tensorcnt(0);
    __syncthreads();
#pragma unroll
    for (int ks = 0; ks < 64; ks += 32) {
      FragAB a[2], bf[4];
      const int m = lane & 15, hi = lane >> 4;
#pragma unroll
      for (int t = 0; t < 2; ++t) {
        const unsigned short* ab = &As[(wrow * 32 + t * 16 + m) * 64 + ks + hi * 8];
        a[t].f4[0] = *(const f4_t*)(ab);
        a[t].f4[1] = *(const f4_t*)(ab + 16);
      }
#pragma unroll
      for (int j = 0; j < 4; ++j) {
        const unsigned short* bb = &Bs[(ks + (lane >> 4) * 8) * 128 + wcol * 64 + j * 16 + (lane & 15)];
#pragma unroll
        for (int s = 0; s < 8; ++s) {
          ((unsigned short*)&bf[j])[s] = bb[s * 128];
          ((unsigned short*)&bf[j])[s + 8] = bb[(s + 16) * 128];
        }
      }
#pragma unroll
      for (int t = 0; t < 2; ++t)
#pragma unroll
        for (int j = 0; j < 4; ++j)
          acc[t][j] = WMMA_BF16(a[t].v, bf[j].v, acc[t][j]);
    }
    __syncthreads();
  }

  const int nlo = lane & 15, mh = lane >> 4;
  if (bn < 512) {
    // fx epilogue
#pragma unroll
    for (int t = 0; t < 2; ++t)
#pragma unroll
      for (int j = 0; j < 4; ++j) {
        const int ng = bn + wcol * 64 + j * 16 + nlo;
        const float bo = bfx[ng];
#pragma unroll
        for (int r = 0; r < 8; ++r) {
          const int mr = bm + wrow * 32 + t * 16 + mh * 8 + r;
          fx_out[(size_t)mr * 512 + ng] = f2bf(acc[t][j][r] + bo);
        }
      }
  } else {
    // softmax epilogue: this wave owns full 64-col head group [colbase, +64)
    const int colbase = bn + wcol * 64 - 512;          // 0..448, 64-aligned
    const int h = colbase >> 6;
    float tp = temperature[h];
    tp = fminf(fmaxf(tp, 0.5f), 5.0f);
    const float it = 1.0f / tp;
    float bcol[4], nsum[4];
#pragma unroll
    for (int j = 0; j < 4; ++j) { bcol[j] = bps[colbase + j * 16 + nlo]; nsum[j] = 0.f; }
#pragma unroll
    for (int t = 0; t < 2; ++t)
#pragma unroll
      for (int r = 0; r < 8; ++r) {
        float v[4];
#pragma unroll
        for (int j = 0; j < 4; ++j) v[j] = (acc[t][j][r] + bcol[j]) * it;
        float mx = fmaxf(fmaxf(v[0], v[1]), fmaxf(v[2], v[3]));
#pragma unroll
        for (int o = 1; o <= 8; o <<= 1) mx = fmaxf(mx, __shfl_xor(mx, o, 32));
        float e[4], sm = 0.f;
#pragma unroll
        for (int j = 0; j < 4; ++j) { e[j] = __expf(v[j] - mx); sm += e[j]; }
#pragma unroll
        for (int o = 1; o <= 8; o <<= 1) sm += __shfl_xor(sm, o, 32);
        const float inv = 1.0f / sm;
        const int mr = bm + wrow * 32 + t * 16 + mh * 8 + r;
#pragma unroll
        for (int j = 0; j < 4; ++j) {
          const float w = e[j] * inv;
          nsum[j] += w;
          sw_out[(size_t)mr * 512 + colbase + j * 16 + nlo] = f2bf(w);
        }
      }
#pragma unroll
    for (int j = 0; j < 4; ++j)
      atomicAdd(&snorm[b * 512 + colbase + j * 16 + nlo], nsum[j]);
  }
}

// ---------------------------------------------------------------------------
// K3: slice_tokens (unnormalized) = sw^T @ fx per (b,h). M=N=64, K=32768
// split into 8 chunks; f32 atomic accumulate. 4 waves; A transposed in LDS.
// ---------------------------------------------------------------------------
__global__ __launch_bounds__(128)
void k3_slice_tokens(const unsigned short* __restrict__ sw,
                     const unsigned short* __restrict__ fx,
                     float* __restrict__ stok)
{
  __shared__ alignas(16) unsigned short As[64 * 32];  // [s][k]
  __shared__ alignas(16) unsigned short Bs[32 * 64];  // [k][d]
  const int tid = threadIdx.x, lane = tid & 31, wave = tid >> 5;
  const int bh = blockIdx.x, b = bh >> 3, h = bh & 7;
  const long nbase = (long)b * 32768 + (long)blockIdx.y * 4096;

  floatx8 acc[4];
#pragma unroll
  for (int j = 0; j < 4; ++j)
#pragma unroll
    for (int r = 0; r < 8; ++r) acc[j][r] = 0.f;

  for (int it = 0; it < 128; ++it) {
    const long trow = nbase + (long)it * 32;
#pragma unroll
    for (int i = 0; i < 4; ++i) {
      int idx = tid + i * 128;
      int tk = idx >> 4, c4 = (idx & 15) << 2;
      u2_t v = *(const u2_t*)(sw + (size_t)(trow + tk) * 512 + h * 64 + c4);
      As[(c4 + 0) * 32 + tk] = (unsigned short)(v.x & 0xFFFFu);
      As[(c4 + 1) * 32 + tk] = (unsigned short)(v.x >> 16);
      As[(c4 + 2) * 32 + tk] = (unsigned short)(v.y & 0xFFFFu);
      As[(c4 + 3) * 32 + tk] = (unsigned short)(v.y >> 16);
      *(u2_t*)(&Bs[tk * 64 + c4]) =
          *(const u2_t*)(fx + (size_t)(trow + tk) * 512 + h * 64 + c4);
    }
    __syncthreads();
    FragAB a, bf[4];
    const int m = lane & 15, hi = lane >> 4;
    const unsigned short* ab = &As[(wave * 16 + m) * 32 + hi * 8];
    a.f4[0] = *(const f4_t*)(ab);
    a.f4[1] = *(const f4_t*)(ab + 16);
#pragma unroll
    for (int j = 0; j < 4; ++j) {
      const unsigned short* bb = &Bs[((lane >> 4) * 8) * 64 + j * 16 + (lane & 15)];
#pragma unroll
      for (int s = 0; s < 8; ++s) {
        ((unsigned short*)&bf[j])[s] = bb[s * 64];
        ((unsigned short*)&bf[j])[s + 8] = bb[(s + 16) * 64];
      }
    }
#pragma unroll
    for (int j = 0; j < 4; ++j) acc[j] = WMMA_BF16(a.v, bf[j].v, acc[j]);
    __syncthreads();
  }
  float* dst = stok + (size_t)bh * 4096;
  const int nlo = lane & 15, mh = lane >> 4;
#pragma unroll
  for (int j = 0; j < 4; ++j)
#pragma unroll
    for (int r = 0; r < 8; ++r)
      atomicAdd(&dst[(wave * 16 + mh * 8 + r) * 64 + j * 16 + nlo], acc[j][r]);
}

// ---------------------------------------------------------------------------
// K4: per (b,h): normalize tokens, qkv, 64x64 SDPA, M2 = out_slice @ Wout_h.
// ---------------------------------------------------------------------------
__global__ __launch_bounds__(256)
void k4_attn(const float* __restrict__ stok, const float* __restrict__ snorm,
             const float* __restrict__ Wqkv, const float* __restrict__ Wout,
             unsigned short* __restrict__ M2)
{
  __shared__ float tok[64 * 64];
  __shared__ float qkv[64 * 192];
  const int tid = threadIdx.x;
  const int bh = blockIdx.x, b = bh >> 3, h = bh & 7;
  const float* st = stok + (size_t)bh * 4096;
  for (int i = tid; i < 4096; i += 256) {
    int s = i >> 6;
    tok[i] = st[i] / snorm[b * 512 + h * 64 + s];
  }
  __syncthreads();
  for (int i = tid; i < 64 * 192; i += 256) {
    int s = i / 192, c = i % 192;
    float a = 0.f;
#pragma unroll 8
    for (int d = 0; d < 64; ++d) a += tok[s * 64 + d] * Wqkv[d * 192 + c];
    qkv[i] = a;
  }
  __syncthreads();
  for (int i = tid; i < 4096; i += 256) {
    int q = i >> 6, k = i & 63;
    float a = 0.f;
#pragma unroll 8
    for (int d = 0; d < 64; ++d) a += qkv[q * 192 + d] * qkv[k * 192 + 64 + d];
    tok[i] = a * 0.125f;
  }
  __syncthreads();
  if (tid < 64) {
    float mx = -1e30f;
    for (int k = 0; k < 64; ++k) mx = fmaxf(mx, tok[tid * 64 + k]);
    float sm = 0.f;
    for (int k = 0; k < 64; ++k) { float e = __expf(tok[tid * 64 + k] - mx); tok[tid * 64 + k] = e; sm += e; }
    float inv = 1.f / sm;
    for (int k = 0; k < 64; ++k) tok[tid * 64 + k] *= inv;
  }
  __syncthreads();
  for (int i = tid; i < 4096; i += 256) {
    int s = i >> 6, d = i & 63;
    float a = 0.f;
#pragma unroll 8
    for (int k = 0; k < 64; ++k) a += tok[s * 64 + k] * qkv[k * 192 + 128 + d];
    qkv[s * 192 + d] = a;
  }
  __syncthreads();
  unsigned short* m2 = M2 + (size_t)b * 512 * 512 + (size_t)h * 64 * 512;
  for (int i = tid; i < 64 * 512; i += 256) {
    int s = i >> 9, c = i & 511;
    float a = 0.f;
#pragma unroll 8
    for (int d = 0; d < 64; ++d) a += qkv[s * 192 + d] * Wout[(size_t)(h * 64 + d) * 512 + c];
    m2[(size_t)s * 512 + c] = f2bf(a);
  }
}

// ---------------------------------------------------------------------------
// K5: out = SW[b] @ M2[b] + bout.  TDM double-buffered tile pipeline:
// wave 0 prefetches tile k+1 via tensor_load_to_lds while all waves WMMA
// tile k; tensor ops of one wave complete in order -> s_wait_tensorcnt(2).
// ---------------------------------------------------------------------------
__global__ __launch_bounds__(256)
void k5_gemm_out(const unsigned short* __restrict__ sw,
                 const unsigned short* __restrict__ M2,
                 const float* __restrict__ bout, float* __restrict__ out)
{
  __shared__ alignas(16) unsigned short As[2][128 * 64];
  __shared__ alignas(16) unsigned short Bs[2][64 * 128];
  const int tid = threadIdx.x, lane = tid & 31, wave = tid >> 5;
  const int wrow = wave & 3, wcol = wave >> 2;
  const int bm = blockIdx.x * 128;
  const int bn = blockIdx.y * 128;
  const int b  = bm >> 15;
  const unsigned short* Arow = sw + (size_t)bm * 512;
  const unsigned short* Brow = M2 + (size_t)b * 512 * 512 + bn;

  floatx8 acc[2][4];
#pragma unroll
  for (int t = 0; t < 2; ++t)
#pragma unroll
    for (int j = 0; j < 4; ++j)
#pragma unroll
      for (int r = 0; r < 8; ++r) acc[t][j][r] = 0.f;

  if (wave == 0) {
    tdm_load_tile_bf16(LDS_OFF(As[0]), Arow, 64, 128, 512);
    tdm_load_tile_bf16(LDS_OFF(Bs[0]), Brow, 128, 64, 512);
  }
  int buf = 0;
  for (int kt = 0; kt < 8; ++kt) {
    if (wave == 0) {
      if (kt < 7) {
        const int k1 = (kt + 1) * 64;
        tdm_load_tile_bf16(LDS_OFF(As[buf ^ 1]), Arow + k1, 64, 128, 512);
        tdm_load_tile_bf16(LDS_OFF(Bs[buf ^ 1]), Brow + (size_t)k1 * 512, 128, 64, 512);
        __builtin_amdgcn_s_wait_tensorcnt(2);   // tile kt done, kt+1 in flight
      } else {
        __builtin_amdgcn_s_wait_tensorcnt(0);
      }
    }
    __syncthreads();
    const unsigned short* Ab = As[buf];
    const unsigned short* Bb = Bs[buf];
#pragma unroll
    for (int ks = 0; ks < 64; ks += 32) {
      FragAB a[2], bf[4];
      const int m = lane & 15, hi = lane >> 4;
#pragma unroll
      for (int t = 0; t < 2; ++t) {
        const unsigned short* ab = &Ab[(wrow * 32 + t * 16 + m) * 64 + ks + hi * 8];
        a[t].f4[0] = *(const f4_t*)(ab);
        a[t].f4[1] = *(const f4_t*)(ab + 16);
      }
#pragma unroll
      for (int j = 0; j < 4; ++j) {
        const unsigned short* bb = &Bb[(ks + (lane >> 4) * 8) * 128 + wcol * 64 + j * 16 + (lane & 15)];
#pragma unroll
        for (int s = 0; s < 8; ++s) {
          ((unsigned short*)&bf[j])[s] = bb[s * 128];
          ((unsigned short*)&bf[j])[s + 8] = bb[(s + 16) * 128];
        }
      }
#pragma unroll
      for (int t = 0; t < 2; ++t)
#pragma unroll
        for (int j = 0; j < 4; ++j)
          acc[t][j] = WMMA_BF16(a[t].v, bf[j].v, acc[t][j]);
    }
    __syncthreads();   // all done reading buf before wave0 overwrites next iter
    buf ^= 1;
  }
  const int nlo = lane & 15, mh = lane >> 4;
#pragma unroll
  for (int t = 0; t < 2; ++t)
#pragma unroll
    for (int j = 0; j < 4; ++j) {
      const int ng = bn + wcol * 64 + j * 16 + nlo;
      const float bo = bout[ng];
#pragma unroll
      for (int r = 0; r < 8; ++r) {
        const int mr = bm + wrow * 32 + t * 16 + mh * 8 + r;
        out[(size_t)mr * 512 + ng] = acc[t][j][r] + bo;
      }
    }
}

// ---------------------------------------------------------------------------
extern "C" void kernel_launch(void* const* d_in, const int* in_sizes, int n_in,
                              void* d_out, int out_size, void* d_ws, size_t ws_size,
                              hipStream_t stream) {
  (void)in_sizes; (void)n_in; (void)out_size; (void)ws_size;
  const float* x    = (const float*)d_in[0];
  const float* Wx   = (const float*)d_in[1];
  const float* bx   = (const float*)d_in[2];
  const float* Wfx  = (const float*)d_in[3];
  const float* bfx  = (const float*)d_in[4];
  const float* Wsl  = (const float*)d_in[5];
  const float* bsl  = (const float*)d_in[6];
  const float* Wqkv = (const float*)d_in[7];
  const float* Wout = (const float*)d_in[8];
  const float* bout = (const float*)d_in[9];
  const float* temp = (const float*)d_in[10];

  char* ws = (char*)d_ws;
  unsigned short* fx_bf  = (unsigned short*)(ws + 0);             //  64 MB
  unsigned short* sw_bf  = (unsigned short*)(ws + 67108864ULL);   //  64 MB
  unsigned short* Wfx_bf = (unsigned short*)(ws + 134217728ULL);  // 512 KB
  unsigned short* Wps_bf = (unsigned short*)(ws + 134742016ULL);  // 512 KB
  float*          bps    = (float*)(ws + 135266304ULL);           //   2 KB
  float*          snorm  = (float*)(ws + 135268352ULL);           //   4 KB
  float*          stok   = (float*)(ws + 135272448ULL);           // 256 KB
  unsigned short* M2     = (unsigned short*)(ws + 135534592ULL);  //   1 MB

  k0_prep<<<512, 512, 0, stream>>>(Wx, Wfx, Wsl, bx, bsl, Wfx_bf, Wps_bf, bps, snorm, stok);
  k1_gemm_fused<<<dim3(512, 8), 256, 0, stream>>>(x, Wfx_bf, Wps_bf, bfx, bps, temp,
                                                  fx_bf, sw_bf, snorm);
  k3_slice_tokens<<<dim3(16, 8), 128, 0, stream>>>(sw_bf, fx_bf, stok);
  k4_attn<<<16, 256, 0, stream>>>(stok, snorm, Wqkv, Wout, M2);
  k5_gemm_out<<<dim3(512, 4), 256, 0, stream>>>(sw_bf, M2, bout, (float*)d_out);
}
